// AttentionGNNLayer_41592463294574
// MI455X (gfx1250) — matrix operand order repaired and checked by reading, not compiled, not run on hardware
//
#include <hip/hip_runtime.h>

#define N_NODES 100000
#define N_EDGES 1600000
#define HALF_E  (N_EDGES / 2)
#define D 32
#define NTILES  (N_EDGES / 16)       // 100000 edge tiles
#define WAVES_PER_BLOCK 8

typedef __attribute__((ext_vector_type(16))) __bf16 v16bf;
typedef __attribute__((ext_vector_type(8)))  __bf16 v8bf;
typedef __attribute__((ext_vector_type(8)))  float  v8f;
typedef __attribute__((ext_vector_type(4)))  float  v4f;

__device__ __forceinline__ float selu_f(float x) {
  const float kScale = 1.0507009873554805f;
  const float kAlpha = 1.6732632423543772f;
  return kScale * (x > 0.0f ? x : kAlpha * (__expf(x) - 1.0f));
}

// ---------------------------------------------------------------------------
// Pack W1 (65x65) and W2 (65x32) into bf16 WMMA B-fragment register order.
// B fragment (32x16 bf16): lane L holds col N=L%16, element j holds row K=(j&7)+8*(L/16)+16*(j/8).
// W1: 5 N-tiles (N padded 65->80) x 3 K-frags (K padded 65->96) = 15 frags.
// W2: 2 N-tiles (N=32)            x 3 K-frags (K padded 65->96) =  6 frags.
// Each fragment = 32 lanes * 16 bf16 = 512 bf16, per-lane contiguous 32B.
// ---------------------------------------------------------------------------
__global__ void prep_weights(const float* __restrict__ W1,
                             const float* __restrict__ W2,
                             __bf16* __restrict__ fW1,
                             __bf16* __restrict__ fW2) {
  int idx = blockIdx.x * blockDim.x + threadIdx.x;
  if (idx < 15 * 512) {
    int f = idx >> 9, r = idx & 511;
    int lanei = r >> 4, j = r & 15;
    int t = f / 3, kf = f % 3;
    int K = kf * 32 + (j & 7) + ((lanei >> 4) << 3) + ((j >> 3) << 4), N = t * 16 + (lanei & 15);
    float v = (K < 65 && N < 65) ? W1[K * 65 + N] : 0.0f;
    fW1[idx] = (__bf16)v;
  } else if (idx < 21 * 512) {
    int i2 = idx - 15 * 512;
    int f = i2 >> 9, r = i2 & 511;
    int lanei = r >> 4, j = r & 15;
    int t = f / 3, kf = f % 3;
    int K = kf * 32 + (j & 7) + ((lanei >> 4) << 3) + ((j >> 3) << 4), N = t * 16 + (lanei & 15);   // N < 32 always
    float v = (K < 65) ? W2[K * 32 + N] : 0.0f;
    fW2[i2] = (__bf16)v;
  }
}

__global__ void zero_out(float* __restrict__ out, int n) {
  int i = blockIdx.x * blockDim.x + threadIdx.x;
  if (i < n) out[i] = 0.0f;
}

__global__ void relu_inplace(float* __restrict__ out, int n) {
  int i = blockIdx.x * blockDim.x + threadIdx.x;
  if (i < n) out[i] = fmaxf(out[i], 0.0f);
}

// ---------------------------------------------------------------------------
// One wave32 per 16-edge tile.
// GEMM1: [recv|send|c] (16x96, bf16) x W1frags -> h1 (16x80) ; selu ; ->LDS
// GEMM2: h1 (16x96 padded, from LDS) x W2frags -> msgs (16x32); selu
// Scatter: global_atomic_add_f32 into out[receiver][:]
// ---------------------------------------------------------------------------
__global__ __launch_bounds__(256) void gnn_edge_wmma(
    const float* __restrict__ h,
    const float* __restrict__ coup,
    const float* __restrict__ b1,
    const float* __restrict__ b2,
    const int* __restrict__ senders,
    const int* __restrict__ receivers,
    const __bf16* __restrict__ fW1,
    const __bf16* __restrict__ fW2,
    float* __restrict__ out) {
  __shared__ __bf16 lds_h1[WAVES_PER_BLOCK][16 * 96];
  __shared__ int    lds_r[WAVES_PER_BLOCK][16];

  const int w    = threadIdx.x >> 5;
  const int lane = threadIdx.x & 31;
  const int row  = lane & 15;      // M row for A frags / N col for C frags
  const int hlf  = lane >> 4;      // lane-group (K/M offset selector)
  const int tile = blockIdx.x * WAVES_PER_BLOCK + w;
  const int e    = tile * 16 + row;

  const int   ridx = receivers[e];
  const int   sidx = senders[e];
  const float cv   = coup[(e < HALF_E) ? e : (e - HALF_E)];
  if (hlf == 0) lds_r[w][row] = ridx;

  // Zero pad columns 80..95 of this wave's h1 tile (K-padding for GEMM2).
  {
    v8bf z = {};
    *(v8bf*)&lds_h1[w][row * 96 + 80 + hlf * 8] = z;
  }

  // --- Build A fragments (16x32 bf16 layout): lane<16 holds M=row,
  //     K = 0..7 (elems 0..7) and 16..23 (elems 8..15); lanes 16..31 get +8.
  const float* hr = h + (long long)ridx * D + hlf * 8;
  const float* hs = h + (long long)sidx * D + hlf * 8;
  v4f r0 = *(const v4f*)(hr + 0),  r1 = *(const v4f*)(hr + 4);
  v4f r2 = *(const v4f*)(hr + 16), r3 = *(const v4f*)(hr + 20);
  v4f s0 = *(const v4f*)(hs + 0),  s1 = *(const v4f*)(hs + 4);
  v4f s2 = *(const v4f*)(hs + 16), s3 = *(const v4f*)(hs + 20);

  v16bf aR, aS, aC;
#pragma unroll
  for (int i = 0; i < 4; ++i) {
    aR[i]      = (__bf16)r0[i];  aR[4 + i]  = (__bf16)r1[i];
    aR[8 + i]  = (__bf16)r2[i];  aR[12 + i] = (__bf16)r3[i];
    aS[i]      = (__bf16)s0[i];  aS[4 + i]  = (__bf16)s1[i];
    aS[8 + i]  = (__bf16)s2[i];  aS[12 + i] = (__bf16)s3[i];
    aC[i] = (__bf16)0.0f;        aC[4 + i]  = (__bf16)0.0f;
    aC[8 + i] = (__bf16)0.0f;    aC[12 + i] = (__bf16)0.0f;
  }
  if (hlf == 0) aC[0] = (__bf16)cv;   // K=64 carries the coupling scalar

  // --- GEMM1: 5 N-tiles x 3 K-frags, f32 accumulate, selu, spill to LDS.
  const __bf16* w1l = fW1 + lane * 16;
#pragma unroll
  for (int t = 0; t < 5; ++t) {
    v16bf bR = *(const v16bf*)(w1l + (t * 3 + 0) * 512);
    v16bf bS = *(const v16bf*)(w1l + (t * 3 + 1) * 512);
    v16bf bC = *(const v16bf*)(w1l + (t * 3 + 2) * 512);
    v8f acc = {};
    acc = __builtin_amdgcn_wmma_f32_16x16x32_bf16(false, aR, false, bR, (short)0, acc, false, false);
    acc = __builtin_amdgcn_wmma_f32_16x16x32_bf16(false, aS, false, bS, (short)0, acc, false, false);
    acc = __builtin_amdgcn_wmma_f32_16x16x32_bf16(false, aC, false, bC, (short)0, acc, false, false);
    const int n = t * 16 + row;
    const float bias = (n < 65) ? b1[n] : 0.0f;   // pad cols -> selu(0)=0
#pragma unroll
    for (int i = 0; i < 8; ++i) {
      lds_h1[w][(i + hlf * 8) * 96 + n] = (__bf16)selu_f(acc[i] + bias);
    }
  }

#if __has_builtin(__builtin_amdgcn_s_wait_dscnt)
  __builtin_amdgcn_s_wait_dscnt(0);
#else
  asm volatile("s_wait_dscnt 0" ::: "memory");
#endif

  // --- Reload h1 in A-fragment layout from LDS (two 16B ds reads per frag).
  v16bf ha[3];
#pragma unroll
  for (int kf = 0; kf < 3; ++kf) {
    const __bf16* p = &lds_h1[w][row * 96 + kf * 32 + hlf * 8];
    v8bf lo = *(const v8bf*)p;
    v8bf hi = *(const v8bf*)(p + 16);
#pragma unroll
    for (int i = 0; i < 8; ++i) { ha[kf][i] = lo[i]; ha[kf][8 + i] = hi[i]; }
  }

  int rdx[8];
#pragma unroll
  for (int i = 0; i < 8; ++i) rdx[i] = lds_r[w][i + hlf * 8];

  // --- GEMM2: 2 N-tiles x 3 K-frags, selu, scatter-add.
  const __bf16* w2l = fW2 + lane * 16;
#pragma unroll
  for (int t = 0; t < 2; ++t) {
    v8f acc = {};
#pragma unroll
    for (int kf = 0; kf < 3; ++kf) {
      v16bf bW = *(const v16bf*)(w2l + (t * 3 + kf) * 512);
      acc = __builtin_amdgcn_wmma_f32_16x16x32_bf16(false, ha[kf], false, bW, (short)0, acc, false, false);
    }
    const float bias = b2[t * 16 + row];
#pragma unroll
    for (int i = 0; i < 8; ++i) {
      float msg = selu_f(acc[i] + bias);
      unsafeAtomicAdd(out + (long long)rdx[i] * D + t * 16 + row, msg);
    }
  }
}

extern "C" void kernel_launch(void* const* d_in, const int* in_sizes, int n_in,
                              void* d_out, int out_size, void* d_ws, size_t ws_size,
                              hipStream_t stream) {
  const float* h     = (const float*)d_in[0];
  const float* coup  = (const float*)d_in[1];
  const float* W1    = (const float*)d_in[2];
  const float* b1    = (const float*)d_in[3];
  const float* W2    = (const float*)d_in[4];
  const float* b2    = (const float*)d_in[5];
  // d_in[6..9] = Wq, bq, Wk, bk: softmax over a size-1 axis is identically 1,
  // so the attention branch is mathematically a no-op and is skipped.
  const int* senders   = (const int*)d_in[10];
  const int* receivers = (const int*)d_in[11];
  float* out = (float*)d_out;

  __bf16* fW1 = (__bf16*)d_ws;            // 15 * 512 bf16 = 15360 B
  __bf16* fW2 = fW1 + 15 * 512;           //  6 * 512 bf16 =  6144 B

  prep_weights<<<(21 * 512 + 255) / 256, 256, 0, stream>>>(W1, W2, fW1, fW2);
  zero_out<<<(N_NODES * D + 255) / 256, 256, 0, stream>>>(out, N_NODES * D);
  gnn_edge_wmma<<<NTILES / WAVES_PER_BLOCK, 256, 0, stream>>>(
      h, coup, b1, b2, senders, receivers, fW1, fW2, out);
  relu_inplace<<<(N_NODES * D + 255) / 256, 256, 0, stream>>>(out, N_NODES * D);
}
